// DecoderWithAttention_27230092656744
// MI455X (gfx1250) — matrix operand order repaired and checked by reading, not compiled, not run on hardware
//
#include <hip/hip_runtime.h>
#include <math.h>

#define Bn   64
#define Pn   196
#define ENCn 2048
#define En   512
#define Hn   512
#define An   512
#define Vn   10000
#define Ln   21
#define Tn   20

typedef __attribute__((ext_vector_type(16))) __bf16       v16bf;
typedef __attribute__((ext_vector_type(8)))  float        v8f;
typedef __attribute__((ext_vector_type(4)))  unsigned int u32x4;

union Frag { v16bf v; u32x4 q[2]; };
union Pk   { unsigned u; __bf16 h[2]; };

#define ASTR 40    // LDS row stride (bf16) for A tile [m][k]; 80B rows, 16B aligned
#define BSTR 40    // LDS row stride (bf16) for B tile stored transposed [n][k]
#define BM   64
#define BN   128

// ---------------------------------------------------------------------------
// WMMA GEMM: C[M,N] = A[M,K] @ B[K,N] (+bias) (+=C if accumulate)
// fp32 in/out, bf16 WMMA with f32 accumulation. M % 64 == 0, K % 32 == 0.
// Block tile 64x128, 8 waves = 4(M) x 2(N), 4 accumulators/wave (16x64).
// Double-buffered LDS staging: next K-tile's global loads overlap WMMAs.
// ---------------------------------------------------------------------------
__global__ __launch_bounds__(256) void gemm_wmma_bf16(
    const float* __restrict__ A, int lda,
    const float* __restrict__ B, int ldb,
    float* __restrict__ C, int ldc,
    const float* __restrict__ bias,
    int N, int K, int accumulate)
{
  __shared__ __align__(16) __bf16 As[2][BM * ASTR];
  __shared__ __align__(16) __bf16 Bs[2][BN * BSTR];

  const int tid   = threadIdx.x;
  const int lane  = tid & 31;
  const int wave  = tid >> 5;
  const int waveM = wave & 3;    // 4 M-subtiles of 16
  const int waveN = wave >> 2;   // 2 N-subtiles of 64
  const int tileM = blockIdx.y * BM;
  const int tileN = blockIdx.x * BN;
  const bool fullN = (tileN + BN) <= N;   // uniform per block

  v8f acc[4];
  #pragma unroll
  for (int j = 0; j < 4; ++j) acc[j] = (v8f){0.f,0.f,0.f,0.f,0.f,0.f,0.f,0.f};

  const int arow  = tid >> 2;        // 0..63   (m in tile)
  const int acol  = (tid & 3) * 8;   // 0,8,16,24
  const int bk2   = (tid >> 4) * 2;  // 0,2,..,30  (k pair in tile)
  const int bn8   = (tid & 15) * 8;  // 0..120     (n group in tile)

  const int khalf = lane >> 4;
  const int l16   = lane & 15;

  // ---- staging: A tile (64x32) and B tile (32x128, transposed, pk-packed) ----
  auto stage = [&](int buf, int kt) {
    {
      const float* Ag = A + (size_t)(tileM + arow) * lda + kt + acol;
      #pragma unroll
      for (int i = 0; i < 8; ++i)
        As[buf][arow * ASTR + acol + i] = (__bf16)Ag[i];
    }
    {
      const float* Bg0 = B + (size_t)(kt + bk2) * ldb + tileN + bn8;
      const float* Bg1 = Bg0 + ldb;
      if (fullN) {
        #pragma unroll
        for (int i = 0; i < 8; ++i) {
          Pk p; p.h[0] = (__bf16)Bg0[i]; p.h[1] = (__bf16)Bg1[i];
          *(unsigned*)&Bs[buf][(bn8 + i) * BSTR + bk2] = p.u;
        }
      } else {
        #pragma unroll
        for (int i = 0; i < 8; ++i) {
          bool ok = (tileN + bn8 + i) < N;
          Pk p; p.h[0] = (__bf16)(ok ? Bg0[i] : 0.f); p.h[1] = (__bf16)(ok ? Bg1[i] : 0.f);
          *(unsigned*)&Bs[buf][(bn8 + i) * BSTR + bk2] = p.u;
        }
      }
    }
  };

  // ---- compute one staged K-tile ----
  auto compute = [&](int buf) {
    Frag fa;
    {
      const __bf16* p = &As[buf][(waveM * 16 + l16) * ASTR];
      fa.q[0] = *(const u32x4*)(p + khalf * 8);
      fa.q[1] = *(const u32x4*)(p + 16 + khalf * 8);
    }
    #pragma unroll
    for (int j = 0; j < 4; ++j) {
      Frag fb;
      const __bf16* p = &Bs[buf][(waveN * 64 + j * 16 + l16) * BSTR];
      fb.q[0] = *(const u32x4*)(p + khalf * 8);
      fb.q[1] = *(const u32x4*)(p + 16 + khalf * 8);
      acc[j] = __builtin_amdgcn_wmma_f32_16x16x32_bf16(false, fa.v, false, fb.v,
                                                       (short)0, acc[j], false, false);
    }
  };

  int buf = 0;
  stage(0, 0);
  __syncthreads();
  for (int kt = 0; kt < K; kt += 32) {
    if (kt + 32 < K) stage(buf ^ 1, kt + 32);   // global loads overlap WMMAs below
    compute(buf);
    __syncthreads();
    buf ^= 1;
  }

  // ---- epilogue: D layout -> VGPR r: lanes0-15 M=r, lanes16-31 M=r+8; N=lane%16
  #pragma unroll
  for (int j = 0; j < 4; ++j) {
    int n = tileN + waveN * 64 + j * 16 + l16;
    if (n < N) {
      float bv = bias ? bias[n] : 0.f;
      #pragma unroll
      for (int r = 0; r < 8; ++r) {
        int m = tileM + waveM * 16 + r + khalf * 8;
        float* cp = C + (size_t)m * ldc + n;
        float outv = acc[j][r] + bv;
        if (accumulate) outv += *cp;
        *cp = outv;
      }
    }
  }
}

// ---------------------------------------------------------------------------
// mean over P: mean_enc[b,e] = (1/P) * sum_p features[b,p,e]
// ---------------------------------------------------------------------------
__global__ void mean_feat(const float* __restrict__ features, float* __restrict__ mean_enc)
{
  int idx = blockIdx.x * blockDim.x + threadIdx.x;      // < Bn*ENCn
  int b = idx >> 11, e = idx & (ENCn - 1);
  const float* f = features + (size_t)b * Pn * ENCn + e;
  float s = 0.f;
  for (int p = 0; p < Pn; ++p) s += f[(size_t)p * ENCn];
  mean_enc[idx] = s * (1.f / (float)Pn);
}

// ---------------------------------------------------------------------------
// scores + softmax: alpha[b,p] = softmax_p( relu(att1[b,p,:] + att2[b,:]) . W_full + b_full )
// one block per batch element; att2/W_full held in per-lane registers
// ---------------------------------------------------------------------------
__global__ __launch_bounds__(256) void attn_scores(
    const float* __restrict__ att1, const float* __restrict__ att2,
    const float* __restrict__ W_full, const float* __restrict__ b_full,
    float* __restrict__ alpha)
{
  __shared__ float s_sc[Pn];
  __shared__ float s_red[256];
  const int b = blockIdx.x, tid = threadIdx.x;
  const int lane = tid & 31, wave = tid >> 5;

  float wf[16], sa[16];
  #pragma unroll
  for (int i = 0; i < 16; ++i) {
    int a = lane + 32 * i;
    wf[i] = W_full[a];
    sa[i] = att2[b * An + a];
  }

  for (int p = wave; p < Pn; p += 8) {
    const float* row = att1 + ((size_t)b * Pn + p) * An;
    float s = 0.f;
    #pragma unroll
    for (int i = 0; i < 16; ++i) {
      float v = row[lane + 32 * i] + sa[i];
      v = v > 0.f ? v : 0.f;
      s += v * wf[i];
    }
    #pragma unroll
    for (int o = 16; o > 0; o >>= 1) s += __shfl_down(s, o, 32);
    if (lane == 0) s_sc[p] = s + b_full[0];
  }
  __syncthreads();

  float v = (tid < Pn) ? s_sc[tid] : -1e30f;
  s_red[tid] = v; __syncthreads();
  for (int o = 128; o > 0; o >>= 1) { if (tid < o) s_red[tid] = fmaxf(s_red[tid], s_red[tid + o]); __syncthreads(); }
  float mx = s_red[0]; __syncthreads();
  float e = (tid < Pn) ? __expf(v - mx) : 0.f;
  s_red[tid] = e; __syncthreads();
  for (int o = 128; o > 0; o >>= 1) { if (tid < o) s_red[tid] += s_red[tid + o]; __syncthreads(); }
  float denom = s_red[0];
  if (tid < Pn) alpha[b * Pn + tid] = e / denom;
}

// ---------------------------------------------------------------------------
// gated attention-weighted sum into concat buffer:
// X[b, E + e] = sigmoid(gate_pre[b,e]) * sum_p alpha[b,p] * features[b,p,e]
// ---------------------------------------------------------------------------
__global__ void attn_awe(const float* __restrict__ alpha, const float* __restrict__ features,
                         const float* __restrict__ gate_pre, float* __restrict__ X)
{
  int idx = blockIdx.x * blockDim.x + threadIdx.x;      // < Bn*ENCn
  int b = idx >> 11, e = idx & (ENCn - 1);
  const float* f  = features + (size_t)b * Pn * ENCn + e;
  const float* al = alpha + b * Pn;
  float s = 0.f;
  for (int p = 0; p < Pn; ++p) s += al[p] * f[(size_t)p * ENCn];
  float g = 1.f / (1.f + __expf(-gate_pre[idx]));
  X[(size_t)b * (En + ENCn) + En + e] = g * s;
}

// ---------------------------------------------------------------------------
// embedding gather for step t: X[b, 0:E] = emb[captions[b,t], :]
// ---------------------------------------------------------------------------
__global__ void gather_emb(const int* __restrict__ captions, const float* __restrict__ emb,
                           float* __restrict__ X, int t)
{
  int idx = blockIdx.x * blockDim.x + threadIdx.x;      // < Bn*En
  int b = idx >> 9, e = idx & (En - 1);
  int tok = captions[b * Ln + t];
  X[(size_t)b * (En + ENCn) + e] = emb[(size_t)tok * En + e];
}

// ---------------------------------------------------------------------------
// LSTM cell (in-place h,c update), biases b_ih+b_hh folded in here
// ---------------------------------------------------------------------------
__device__ __forceinline__ float sigf(float x) { return 1.f / (1.f + __expf(-x)); }

__global__ void lstm_cell(const float* __restrict__ gates,
                          const float* __restrict__ b_ih, const float* __restrict__ b_hh,
                          float* __restrict__ h, float* __restrict__ c)
{
  int idx = blockIdx.x * blockDim.x + threadIdx.x;      // < Bn*Hn
  int b = idx >> 9, j = idx & (Hn - 1);
  const float* g = gates + (size_t)b * 4 * Hn;
  float gi = g[j]          + b_ih[j]          + b_hh[j];
  float gf = g[Hn + j]     + b_ih[Hn + j]     + b_hh[Hn + j];
  float gg = g[2 * Hn + j] + b_ih[2 * Hn + j] + b_hh[2 * Hn + j];
  float go = g[3 * Hn + j] + b_ih[3 * Hn + j] + b_hh[3 * Hn + j];
  float cn = sigf(gf) * c[idx] + sigf(gi) * tanhf(gg);
  float hn = sigf(go) * tanhf(cn);
  c[idx] = cn;
  h[idx] = hn;
}

// ---------------------------------------------------------------------------
static inline void gemm(const float* A, int lda, const float* B, int ldb,
                        float* C, int ldc, const float* bias,
                        int M, int N, int K, int acc, hipStream_t s)
{
  dim3 grid((N + BN - 1) / BN, M / BM);
  gemm_wmma_bf16<<<grid, 256, 0, s>>>(A, lda, B, ldb, C, ldc, bias, N, K, acc);
}

extern "C" void kernel_launch(void* const* d_in, const int* in_sizes, int n_in,
                              void* d_out, int out_size, void* d_ws, size_t ws_size,
                              hipStream_t stream)
{
  const float* features  = (const float*)d_in[0];
  const int*   captions  = (const int*)  d_in[1];
  const float* W_enc_att = (const float*)d_in[2];
  const float* b_enc_att = (const float*)d_in[3];
  const float* W_dec_att = (const float*)d_in[4];
  const float* b_dec_att = (const float*)d_in[5];
  const float* W_full    = (const float*)d_in[6];
  const float* b_full    = (const float*)d_in[7];
  const float* emb       = (const float*)d_in[8];
  const float* W_ih      = (const float*)d_in[9];
  const float* W_hh      = (const float*)d_in[10];
  const float* b_ih      = (const float*)d_in[11];
  const float* b_hh      = (const float*)d_in[12];
  const float* W_init_h  = (const float*)d_in[13];
  const float* b_init_h  = (const float*)d_in[14];
  const float* W_init_c  = (const float*)d_in[15];
  const float* b_init_c  = (const float*)d_in[16];
  const float* W_fbeta   = (const float*)d_in[17];
  const float* b_fbeta   = (const float*)d_in[18];
  const float* W_fc      = (const float*)d_in[19];
  const float* b_fc      = (const float*)d_in[20];
  float* out = (float*)d_out;

  // workspace layout (fp32)
  float* ws = (float*)d_ws;
  float* mean_enc = ws;                          // 64*2048
  float* h        = mean_enc + Bn * ENCn;        // 64*512
  float* c        = h + Bn * Hn;                 // 64*512
  float* att1     = c + Bn * Hn;                 // 64*196*512
  float* att2     = att1 + (size_t)Bn * Pn * An; // 64*512
  float* alpha    = att2 + Bn * An;              // 64*196
  float* gate_pre = alpha + Bn * Pn;             // 64*2048
  float* X        = gate_pre + Bn * ENCn;        // 64*2560
  float* gates    = X + Bn * (En + ENCn);        // 64*2048

  // ---- prologue ----
  mean_feat<<<(Bn * ENCn) / 256, 256, 0, stream>>>(features, mean_enc);
  gemm(mean_enc, ENCn, W_init_h, Hn, h, Hn, b_init_h, Bn, Hn, ENCn, 0, stream);
  gemm(mean_enc, ENCn, W_init_c, Hn, c, Hn, b_init_c, Bn, Hn, ENCn, 0, stream);
  // big GEMM: att1 = features @ W_enc_att + b_enc_att   (12544 x 512 x 2048)
  gemm(features, ENCn, W_enc_att, An, att1, An, b_enc_att, Bn * Pn, An, ENCn, 0, stream);

  // ---- 20-step decode ----
  for (int t = 0; t < Tn; ++t) {
    gemm(h, Hn, W_dec_att, An, att2, An, b_dec_att, Bn, An, Hn, 0, stream);
    attn_scores<<<Bn, 256, 0, stream>>>(att1, att2, W_full, b_full, alpha);
    gemm(h, Hn, W_fbeta, ENCn, gate_pre, ENCn, b_fbeta, Bn, ENCn, Hn, 0, stream);
    attn_awe<<<(Bn * ENCn) / 256, 256, 0, stream>>>(alpha, features, gate_pre, X);
    gather_emb<<<(Bn * En) / 256, 256, 0, stream>>>(captions, emb, X, t);
    gemm(X, En + ENCn, W_ih, 4 * Hn, gates, 4 * Hn, nullptr, Bn, 4 * Hn, En + ENCn, 0, stream);
    gemm(h, Hn, W_hh, 4 * Hn, gates, 4 * Hn, nullptr, Bn, 4 * Hn, Hn, 1, stream);
    lstm_cell<<<(Bn * Hn) / 256, 256, 0, stream>>>(gates, b_ih, b_hh, h, c);
    gemm(h, Hn, W_fc, Vn, out + (size_t)t * Vn, Tn * Vn, b_fc, Bn, Vn, Hn, 0, stream);
  }
}
